// DendriticLayerSiLU_Template_87471303950863
// MI455X (gfx1250) — compile-verified, matrix operand order checked by reading, not run on hardware
//
#include <hip/hip_runtime.h>

// Problem constants (fixed by the reference).
#define IN_F   2048
#define OUT_F  1024
#define WSZ    64
#define NWIN   32
#define NTOK   4096

#define XN (NTOK * IN_F)    // 8388608 elements (x)
#define WN (OUT_F * IN_F)   // 2097152 elements (weights / templates)

typedef __attribute__((ext_vector_type(16))) __bf16 v16bf;
typedef __attribute__((ext_vector_type(8)))  float  v8f;
typedef __attribute__((ext_vector_type(4)))  int    v4i;

typedef __attribute__((address_space(1))) v4i  gv4i;   // global v4i
typedef __attribute__((address_space(3))) v4i  lv4i;   // LDS v4i

union FragU {
    v16bf v;
    uint4 q[2];
};

#if __has_builtin(__builtin_amdgcn_global_load_async_to_lds_b128)
#define USE_ASYNC_LDS 1
#else
#define USE_ASYNC_LDS 0
#endif

// ---------------------------------------------------------------------------
// f32 -> bf16 split helpers (round-to-nearest-even hi, residual lo)
// ---------------------------------------------------------------------------
__device__ __forceinline__ unsigned short f2bf_rne(float f) {
    unsigned u = __float_as_uint(f);
    unsigned r = u + 0x7FFFu + ((u >> 16) & 1u);
    return (unsigned short)(r >> 16);
}
__device__ __forceinline__ float bf2f(unsigned short h) {
    return __uint_as_float(((unsigned)h) << 16);
}

// One-time pre-pass: split an f32 array into bf16 hi/lo arrays (global mem).
// This removes all conversion VALU from the (heavily re-read) GEMM tiles.
__global__ __launch_bounds__(256)
void split_bf16_kernel(const float* __restrict__ src,
                       unsigned short* __restrict__ hi,
                       unsigned short* __restrict__ lo, int n4) {
    int i = blockIdx.x * blockDim.x + threadIdx.x;   // float4 index
    if (i >= n4) return;
    float4 s = ((const float4*)src)[i];
    unsigned short h0 = f2bf_rne(s.x), h1 = f2bf_rne(s.y);
    unsigned short h2 = f2bf_rne(s.z), h3 = f2bf_rne(s.w);
    unsigned short l0 = f2bf_rne(s.x - bf2f(h0));
    unsigned short l1 = f2bf_rne(s.y - bf2f(h1));
    unsigned short l2 = f2bf_rne(s.z - bf2f(h2));
    unsigned short l3 = f2bf_rne(s.w - bf2f(h3));
    uint2 hp, lp;
    hp.x = (unsigned)h0 | ((unsigned)h1 << 16);
    hp.y = (unsigned)h2 | ((unsigned)h3 << 16);
    lp.x = (unsigned)l0 | ((unsigned)l1 << 16);
    lp.y = (unsigned)l2 | ((unsigned)l3 << 16);
    ((uint2*)hi)[i] = hp;
    ((uint2*)lo)[i] = lp;
}

// ---------------------------------------------------------------------------
// GEMM kernel pieces
// ---------------------------------------------------------------------------
#define WMMA_BF16(A, B, C) \
    __builtin_amdgcn_wmma_f32_16x16x32_bf16(false, (A), false, (B), (short)0, (C), false, false)

__device__ __forceinline__ void wait_async() {
#if USE_ASYNC_LDS
#if __has_builtin(__builtin_amdgcn_s_wait_asynccnt)
    __builtin_amdgcn_s_wait_asynccnt(0);
#else
    asm volatile("s_wait_asynccnt 0x0" ::: "memory");
#endif
#endif
}

// Stage one 64x64 bf16 tile (8 KB) from global into LDS.
// 256 threads x 2 b128 transfers each. Async path uses the CDNA5
// global->LDS DMA (tracked by ASYNCcnt); fallback is load+ds_store.
__device__ __forceinline__ void stage_tile(const unsigned short* __restrict__ g,
                                           int gRow0, int k0,
                                           unsigned short* __restrict__ lds, int tid) {
#pragma unroll
    for (int t = 0; t < 2; ++t) {
        int c  = tid + t * 256;      // ushort8 chunk index (0..511)
        int r  = c >> 3;             // row 0..63
        int ko = (c & 7) << 3;       // k offset 0..56
        const unsigned short* gp = g + (size_t)(gRow0 + r) * IN_F + k0 + ko;
        unsigned short*       lp = lds + r * 64 + ko;
#if USE_ASYNC_LDS
        __builtin_amdgcn_global_load_async_to_lds_b128(
            (gv4i*)gp, (lv4i*)lp, 0, 0);
#else
        *(uint4*)lp = *(const uint4*)gp;
#endif
    }
}

// Load a 16x32 bf16 A/B fragment from a row-major [64][64] LDS tile.
// (Identical layout for A of X and B of W^T/T^T.)  Two ds_load_b128.
__device__ __forceinline__ v16bf load_frag(const unsigned short* __restrict__ lds,
                                           int row0, int kk, int lane) {
    int r = row0 + (lane & 15);
    int h = (lane >> 4) & 1;
    FragU f;
    f.q[0] = *(const uint4*)(lds + r * 64 + kk + 8 * h);
    f.q[1] = *(const uint4*)(lds + r * 64 + kk + 16 + 8 * h);
    return f.v;
}

__global__ __launch_bounds__(256)
void dendritic_silu_fused(const unsigned short* __restrict__ xh,
                          const unsigned short* __restrict__ xl,
                          const unsigned short* __restrict__ wh,
                          const unsigned short* __restrict__ wl,
                          const unsigned short* __restrict__ th,
                          const unsigned short* __restrict__ tl,
                          float* __restrict__ out) {
    extern __shared__ unsigned short smem[];   // 2 buffers x 6 tiles x 4096

    const int tid  = threadIdx.x;
    const int lane = tid & 31;
    const int wid  = tid >> 5;
    const int wm   = wid & 3;        // 4 waves along M
    const int wn   = wid >> 2;       // 2 waves along N
    const int mBase = wm * 16;
    const int nBase = wn * 32;

    const int rowBlk = blockIdx.x * 64;   // token block
    const int colBlk = blockIdx.y * 64;   // unit block

    auto tp = [&](int b, int t) -> unsigned short* {
        return smem + (((b * 6) + t) << 12);   // 4096 ushorts per tile
    };
    auto stage_window = [&](int b, int w) {
        const int k0 = w * WSZ;
        stage_tile(xh, rowBlk, k0, tp(b, 0), tid);
        stage_tile(xl, rowBlk, k0, tp(b, 1), tid);
        stage_tile(wh, colBlk, k0, tp(b, 2), tid);
        stage_tile(wl, colBlk, k0, tp(b, 3), tid);
        stage_tile(th, colBlk, k0, tp(b, 4), tid);
        stage_tile(tl, colBlk, k0, tp(b, 5), tid);
    };

    v8f lin[2] = { {0,0,0,0,0,0,0,0}, {0,0,0,0,0,0,0,0} };  // x @ W^T
    v8f Zac[2] = { {0,0,0,0,0,0,0,0}, {0,0,0,0,0,0,0,0} };  // sum exp(|a|)
    v8f Sac[2] = { {0,0,0,0,0,0,0,0}, {0,0,0,0,0,0,0,0} };  // sum exp(|a|)*a

    stage_window(0, 0);   // prologue: kick off window 0

    for (int w = 0; w < NWIN; ++w) {
        const int b = w & 1;
        wait_async();        // our stage(w) transfers have landed
        __syncthreads();     // everyone's landed; prev readers done

        if (w + 1 < NWIN) stage_window(1 - b, w + 1);   // overlap next copy

        const unsigned short* cxh = tp(b, 0);
        const unsigned short* cxl = tp(b, 1);
        const unsigned short* cwh = tp(b, 2);
        const unsigned short* cwl = tp(b, 3);
        const unsigned short* cth = tp(b, 4);
        const unsigned short* ctl = tp(b, 5);

        v8f aacc[2] = { {0,0,0,0,0,0,0,0}, {0,0,0,0,0,0,0,0} };

#pragma unroll
        for (int kk = 0; kk < WSZ; kk += 32) {
            v16bf aH = load_frag(cxh, mBase, kk, lane);
            v16bf aL = load_frag(cxl, mBase, kk, lane);
#pragma unroll
            for (int nt = 0; nt < 2; ++nt) {
                const int nb = nBase + nt * 16;
                // lin += x * W^T  (bf16x3 ~ fp32 accuracy)
                v16bf bH = load_frag(cwh, nb, kk, lane);
                v16bf bL = load_frag(cwl, nb, kk, lane);
                lin[nt] = WMMA_BF16(aH, bH, lin[nt]);
                lin[nt] = WMMA_BF16(aH, bL, lin[nt]);
                lin[nt] = WMMA_BF16(aL, bH, lin[nt]);
                // a_w += x * T^T (this window only)
                v16bf cH = load_frag(cth, nb, kk, lane);
                v16bf cL = load_frag(ctl, nb, kk, lane);
                aacc[nt] = WMMA_BF16(aH, cH, aacc[nt]);
                aacc[nt] = WMMA_BF16(aH, cL, aacc[nt]);
                aacc[nt] = WMMA_BF16(aL, cH, aacc[nt]);
            }
        }

        // Online softmax sums (TAU=1; |a| ~ 0.2, no max-shift needed).
#pragma unroll
        for (int nt = 0; nt < 2; ++nt) {
#pragma unroll
            for (int e = 0; e < 8; ++e) {
                float a = aacc[nt][e];
                float p = __expf(fabsf(a));
                Zac[nt][e] += p;
                Sac[nt][e] += p * a;
            }
        }
    }

    // Epilogue: g = S/Z, gate = silu(g), out = gate * lin.
    const int half = (lane >> 4) & 1;
#pragma unroll
    for (int nt = 0; nt < 2; ++nt) {
        const int gcol = colBlk + nBase + nt * 16 + (lane & 15);
#pragma unroll
        for (int e = 0; e < 8; ++e) {
            const int grow = rowBlk + mBase + e + 8 * half;
            float g    = Sac[nt][e] / Zac[nt][e];
            float gate = g / (1.0f + __expf(-g));
            out[(size_t)grow * OUT_F + gcol] = gate * lin[nt][e];
        }
    }
}

// ---------------------------------------------------------------------------
extern "C" void kernel_launch(void* const* d_in, const int* in_sizes, int n_in,
                              void* d_out, int out_size, void* d_ws, size_t ws_size,
                              hipStream_t stream) {
    (void)in_sizes; (void)n_in; (void)out_size; (void)ws_size;
    const float* x    = (const float*)d_in[0];   // [4096, 2048]
    const float* tmpl = (const float*)d_in[1];   // [1024, 2048]
    const float* wgt  = (const float*)d_in[2];   // [1024, 2048]
    float* out = (float*)d_out;                  // [4096, 1024]

    // Workspace layout (ushort units): xh | xl | wh | wl | th | tl  (~48 MB)
    unsigned short* ws = (unsigned short*)d_ws;
    unsigned short* xh = ws;
    unsigned short* xl = xh + XN;
    unsigned short* wh = xl + XN;
    unsigned short* wl = wh + WN;
    unsigned short* th = wl + WN;
    unsigned short* tl = th + WN;

    split_bf16_kernel<<<dim3((XN / 4 + 255) / 256), dim3(256), 0, stream>>>(x,    xh, xl, XN / 4);
    split_bf16_kernel<<<dim3((WN / 4 + 255) / 256), dim3(256), 0, stream>>>(wgt,  wh, wl, WN / 4);
    split_bf16_kernel<<<dim3((WN / 4 + 255) / 256), dim3(256), 0, stream>>>(tmpl, th, tl, WN / 4);

    dim3 grid(NTOK / 64, OUT_F / 64);            // (64, 16)
    dim3 block(256);
    size_t smem = (size_t)2 * 6 * 4096 * sizeof(unsigned short);  // 96 KB
    dendritic_silu_fused<<<grid, block, smem, stream>>>(xh, xl, wh, wl, th, tl, out);
}
